// ST_Tokenizer_39651138076928
// MI455X (gfx1250) — compile-verified, matrix operand order checked by reading, not run on hardware
//
#include <hip/hip_runtime.h>

// Problem constants (match reference)
#define NN    20000
#define FF    6
#define EE    160000
#define TT    24
#define DDIM  128
#define EMBD  128
#define HH    2
#define CCH   128
#define VOCAB 31

typedef __attribute__((ext_vector_type(16))) _Float16 v16h;
typedef __attribute__((ext_vector_type(8)))  float    v8f;

// ---- helpers -------------------------------------------------------------

__device__ __forceinline__ unsigned fenc(float f) {
  unsigned u = __float_as_uint(f);
  return (u & 0x80000000u) ? ~u : (u | 0x80000000u);
}
__device__ __forceinline__ float fdec(unsigned e) {
  return __uint_as_float((e & 0x80000000u) ? (e ^ 0x80000000u) : ~e);
}
__device__ __forceinline__ float lrelu(float x) { return x > 0.f ? x : 0.2f * x; }

// ---- embedding gather: B0[n, f*128+e] = tables[f][sf[n,f]][e] ------------

__global__ void k_gather_emb(const int* __restrict__ sf,
                             const float* __restrict__ tables,
                             float* __restrict__ out) {
  int t = blockIdx.x * blockDim.x + threadIdx.x;
  if (t >= NN * FF * EMBD) return;
  int e    = t & (EMBD - 1);
  int rest = t >> 7;
  int f = rest % FF;
  int n = rest / FF;
  int idx = sf[n * FF + f];
  out[(size_t)n * (FF * EMBD) + f * EMBD + e] =
      tables[((size_t)f * VOCAB + idx) * EMBD + e];
}

// ---- pack f32 weight [K x Nw] (or transposed) into WMMA B fragments -------
// Fragment layout: per (kchunk, ntile, lane) 16 contiguous halfs, where
// halfs 0..7  = B[kc*32 + 8g + 0..7][n]
// halfs 8..15 = B[kc*32 + 16 + 8g + 0..7][n],  g = lane>>4, n = nt*16 + lane&15

__global__ void k_pack_b(const float* __restrict__ B, int K, int Nw, int ld,
                         int transpose, _Float16* __restrict__ out) {
  int t = blockIdx.x * blockDim.x + threadIdx.x;
  int nTiles = Nw >> 4;
  int total = (K >> 5) * nTiles * 32;
  if (t >= total) return;
  int lane = t & 31;
  int tile = t >> 5;
  int nt = tile % nTiles;
  int kc = tile / nTiles;
  int n = nt * 16 + (lane & 15);
  int g = lane >> 4;
  _Float16* o = out + (size_t)t * 16;
#pragma unroll
  for (int e = 0; e < 16; ++e) {
    int k = kc * 32 + ((e < 8) ? (8 * g + e) : (16 + 8 * g + (e - 8)));
    float v = transpose ? B[(size_t)n * ld + k] : B[(size_t)k * ld + n];
    o[e] = (_Float16)v;
  }
}

// ---- WMMA GEMM: C[M x Nw] = act(A[M x K](f32) * Bpacked + bias) ----------
// One wave -> one 16 x (NT*16) strip: A fragment loaded ONCE per K-chunk,
// NT WMMAs against NT B fragments (register-blocked over N).
// Block = 8 waves = 128 rows. grid = (ceil(M/128), Nw/(NT*16)).
// Epilogue goes through LDS so the big output stream is written as
// fully-coalesced 512B float4 row bursts.

template <int ACT, int NT>
__global__ void k_gemm(const float* __restrict__ A, int lda, int M, int K,
                       const v16h* __restrict__ Bp, int nTilesTotal,
                       const float* __restrict__ bias,
                       float* __restrict__ C, int ldc, int coff) {
  __shared__ float tile[8 * 16 * (NT * 16)];

  int lane = threadIdx.x & 31;
  int wave = threadIdx.x >> 5;
  int ntBase = blockIdx.y * NT;
  int mbase = (blockIdx.x * 8 + wave) * 16;
  int g = lane >> 4;
  int mA = mbase + (lane & 15);
  const float* arow = A + (size_t)(mA < M ? mA : 0) * lda;

  v8f acc[NT];
#pragma unroll
  for (int j = 0; j < NT; ++j) acc[j] = (v8f){0.f, 0.f, 0.f, 0.f, 0.f, 0.f, 0.f, 0.f};

  int kch = K >> 5;
  for (int kc = 0; kc < kch; ++kc) {
    int k0 = kc * 32 + 8 * g;
    float4 x0 = *(const float4*)(arow + k0);
    float4 x1 = *(const float4*)(arow + k0 + 4);
    float4 x2 = *(const float4*)(arow + k0 + 16);
    float4 x3 = *(const float4*)(arow + k0 + 20);
    __builtin_prefetch(arow + k0 + 32, 0, 1);
    v16h a;
    a[0]  = (_Float16)x0.x; a[1]  = (_Float16)x0.y; a[2]  = (_Float16)x0.z; a[3]  = (_Float16)x0.w;
    a[4]  = (_Float16)x1.x; a[5]  = (_Float16)x1.y; a[6]  = (_Float16)x1.z; a[7]  = (_Float16)x1.w;
    a[8]  = (_Float16)x2.x; a[9]  = (_Float16)x2.y; a[10] = (_Float16)x2.z; a[11] = (_Float16)x2.w;
    a[12] = (_Float16)x3.x; a[13] = (_Float16)x3.y; a[14] = (_Float16)x3.z; a[15] = (_Float16)x3.w;
#pragma unroll
    for (int j = 0; j < NT; ++j) {
      v16h b = Bp[((size_t)kc * nTilesTotal + ntBase + j) * 32 + lane];
      acc[j] = __builtin_amdgcn_wmma_f32_16x16x32_f16(false, a, false, b,
                                                      (short)0, acc[j], false, false);
    }
  }

  // --- epilogue: WMMA C-layout -> LDS -> coalesced float4 row stores ------
  float* my = tile + wave * 16 * (NT * 16);
  int col0 = lane & 15;
#pragma unroll
  for (int j = 0; j < NT; ++j) {
#pragma unroll
    for (int r = 0; r < 8; ++r) {
      my[(g * 8 + r) * (NT * 16) + j * 16 + col0] = acc[j][r];
    }
  }
  __asm__ volatile("" ::: "memory");  // same-wave DS ops are in-order

  int ncLocal = lane * 4;                     // 0..124 (NT*16 == 128 columns)
  int ncGlob = ntBase * 16 + ncLocal;
  float4 bv;
  if (bias) bv = *(const float4*)(bias + ncGlob);
  else      bv = make_float4(0.f, 0.f, 0.f, 0.f);
#pragma unroll
  for (int r = 0; r < 16; ++r) {
    int mr = mbase + r;
    if (mr < M) {
      float4 v = *(float4*)(my + r * (NT * 16) + ncLocal);
      v.x += bv.x; v.y += bv.y; v.z += bv.z; v.w += bv.w;
      if (ACT == 1) {
        v.x = fmaxf(v.x, 0.f); v.y = fmaxf(v.y, 0.f);
        v.z = fmaxf(v.z, 0.f); v.w = fmaxf(v.w, 0.f);
      }
      *(float4*)(C + (size_t)mr * ldc + coff + ncGlob) = v;
    }
  }
}

// ---- GAT attention pieces -------------------------------------------------

__global__ void k_attn(const float* __restrict__ xh,
                       const float* __restrict__ asrc,
                       const float* __restrict__ adst,
                       float* __restrict__ a_s, float* __restrict__ a_d,
                       unsigned* __restrict__ menc) {
  int t = blockIdx.x * blockDim.x + threadIdx.x;
  if (t >= NN * HH) return;
  int h = t & 1, n = t >> 1;
  const float* x = xh + (size_t)n * (HH * CCH) + h * CCH;
  const float* ws = asrc + h * CCH;
  const float* wd = adst + h * CCH;
  float ss = 0.f, sd = 0.f;
  for (int c = 0; c < CCH; ++c) { float v = x[c]; ss += v * ws[c]; sd += v * wd[c]; }
  a_s[t] = ss;
  a_d[t] = sd;
  menc[t] = fenc(lrelu(ss + sd));  // self-loop seeds the running max
}

__global__ void k_edge_max(const int* __restrict__ src, const int* __restrict__ dst,
                           const float* __restrict__ a_s, const float* __restrict__ a_d,
                           unsigned* __restrict__ menc) {
  int t = blockIdx.x * blockDim.x + threadIdx.x;
  if (t >= EE * HH) return;
  int h = t & 1, e = t >> 1;
  int s = src[e], d = dst[e];
  float alpha = lrelu(a_s[s * 2 + h] + a_d[d * 2 + h]);
  atomicMax(&menc[d * 2 + h], fenc(alpha));
}

__global__ void k_node_z(const float* __restrict__ a_s, const float* __restrict__ a_d,
                         const unsigned* __restrict__ menc, float* __restrict__ z) {
  int t = blockIdx.x * blockDim.x + threadIdx.x;
  if (t >= NN * HH) return;
  z[t] = __expf(lrelu(a_s[t] + a_d[t]) - fdec(menc[t]));  // self-loop term
}

__global__ void k_edge_z(const int* __restrict__ src, const int* __restrict__ dst,
                         const float* __restrict__ a_s, const float* __restrict__ a_d,
                         const unsigned* __restrict__ menc, float* __restrict__ z) {
  int t = blockIdx.x * blockDim.x + threadIdx.x;
  if (t >= EE * HH) return;
  int h = t & 1, e = t >> 1;
  int s = src[e], d = dst[e];
  int dh = d * 2 + h;
  float alpha = lrelu(a_s[s * 2 + h] + a_d[dh]);
  atomicAdd(&z[dh], __expf(alpha - fdec(menc[dh])));
}

__global__ void k_agg_init(const float* __restrict__ xh,
                           const float* __restrict__ a_s, const float* __restrict__ a_d,
                           const unsigned* __restrict__ menc, const float* __restrict__ z,
                           float* __restrict__ out) {
  int t = blockIdx.x * blockDim.x + threadIdx.x;
  if (t >= NN * HH * CCH) return;
  int c = t & 255;
  int n = t >> 8;
  int nh = n * 2 + (c >> 7);
  float w = __expf(lrelu(a_s[nh] + a_d[nh]) - fdec(menc[nh])) / (z[nh] + 1e-16f);
  out[t] = w * xh[t];  // self-loop contribution; also initializes accumulator
}

__global__ void k_edge_agg(const int* __restrict__ src, const int* __restrict__ dst,
                           const float* __restrict__ xh,
                           const float* __restrict__ a_s, const float* __restrict__ a_d,
                           const unsigned* __restrict__ menc, const float* __restrict__ z,
                           float* __restrict__ out) {
  int t = blockIdx.x * blockDim.x + threadIdx.x;
  if (t >= EE * 8) return;
  int e = t >> 3;
  int ch = (t & 7) << 5;  // 32-channel chunk
  int h = ch >> 7;
  int s = src[e], d = dst[e];
  int dh = d * 2 + h, sh = s * 2 + h;
  float alpha = lrelu(a_s[sh] + a_d[dh]);
  float w = __expf(alpha - fdec(menc[dh])) / (z[dh] + 1e-16f);
  const float* xs = xh + (size_t)s * 256 + ch;
  float* o = out + (size_t)d * 256 + ch;
#pragma unroll
  for (int i = 0; i < 32; ++i) atomicAdd(&o[i], w * xs[i]);
}

__global__ void k_elu_bias(const float* __restrict__ in, const float* __restrict__ b,
                           float* __restrict__ out) {
  int t = blockIdx.x * blockDim.x + threadIdx.x;
  if (t >= NN * 256) return;
  float v = in[t] + b[t & 255];
  out[t] = v > 0.f ? v : (__expf(v) - 1.f);
}

__global__ void k_mean_bias(const float* __restrict__ in, const float* __restrict__ b,
                            float* __restrict__ out) {
  int t = blockIdx.x * blockDim.x + threadIdx.x;
  if (t >= NN * CCH) return;
  int c = t & 127, n = t >> 7;
  out[t] = 0.5f * (in[(size_t)n * 256 + c] + in[(size_t)n * 256 + 128 + c]) + b[c];
}

// ---- output assembly ------------------------------------------------------

__global__ void k_broadcast_static(const float* __restrict__ st, float* __restrict__ out) {
  int t = blockIdx.x * blockDim.x + threadIdx.x;
  if (t >= NN * TT * 32) return;               // float4 granularity
  int c4 = t & 31;
  int nt = t >> 5;
  int n = nt / TT;
  const float4* s = (const float4*)(st + (size_t)n * 128);
  ((float4*)(out + (size_t)nt * 256))[c4] = s[c4];
}

__global__ void k_special(float* __restrict__ out) {
  int t = blockIdx.x * blockDim.x + threadIdx.x;
  if (t >= 4 * TT * 256) return;
  int i = t / (TT * 256);
  out[(size_t)NN * TT * 256 + t] = (float)i;
}

// ---- launcher -------------------------------------------------------------

extern "C" void kernel_launch(void* const* d_in, const int* in_sizes, int n_in,
                              void* d_out, int out_size, void* d_ws, size_t ws_size,
                              hipStream_t stream) {
  (void)in_sizes; (void)n_in; (void)out_size; (void)ws_size;

  const int*   sf      = (const int*)d_in[0];
  const int*   ei      = (const int*)d_in[1];
  const int*   src     = ei;
  const int*   dst     = ei + EE;
  const float* dynamic = (const float*)d_in[3];
  const float* tables  = (const float*)d_in[4];
  const float* m1w1 = (const float*)d_in[5];  const float* m1b1 = (const float*)d_in[6];
  const float* m1w2 = (const float*)d_in[7];  const float* m1b2 = (const float*)d_in[8];
  const float* g1w  = (const float*)d_in[9];  const float* g1as = (const float*)d_in[10];
  const float* g1ad = (const float*)d_in[11]; const float* g1b  = (const float*)d_in[12];
  const float* g2w  = (const float*)d_in[13]; const float* g2as = (const float*)d_in[14];
  const float* g2ad = (const float*)d_in[15]; const float* g2b  = (const float*)d_in[16];
  const float* m2w1 = (const float*)d_in[17]; const float* m2b1 = (const float*)d_in[18];
  const float* m2w2 = (const float*)d_in[19]; const float* m2b2 = (const float*)d_in[20];
  const float* cw   = (const float*)d_in[21]; const float* cb   = (const float*)d_in[22];
  float* out = (float*)d_out;

  char* ws = (char*)d_ws;
  size_t off = 0;
  auto alloc = [&](size_t bytes) -> void* {
    void* p = ws + off;
    off += (bytes + 255) & ~(size_t)255;
    return p;
  };

  float*    B0   = (float*)alloc((size_t)NN * 768 * 4);  // emb -> h2 -> mlp2 mid
  float*    B1   = (float*)alloc((size_t)NN * 256 * 4);  // mlp1 mid -> agg out -> static
  float*    B2   = (float*)alloc((size_t)NN * 256 * 4);  // h1 -> h3
  float*    B3   = (float*)alloc((size_t)NN * 256 * 4);  // xh1 / xh2
  float*    aS   = (float*)alloc((size_t)NN * 2 * 4);
  float*    aD   = (float*)alloc((size_t)NN * 2 * 4);
  unsigned* menc = (unsigned*)alloc((size_t)NN * 2 * 4);
  float*    zz   = (float*)alloc((size_t)NN * 2 * 4);
  _Float16* p_m1w1 = (_Float16*)alloc(768 * 128 * 2);
  _Float16* p_m1w2 = (_Float16*)alloc(128 * 128 * 2);
  _Float16* p_g1w  = (_Float16*)alloc(128 * 256 * 2);
  _Float16* p_g2w  = (_Float16*)alloc(256 * 256 * 2);
  _Float16* p_m2w1 = (_Float16*)alloc(128 * 128 * 2);
  _Float16* p_m2w2 = (_Float16*)alloc(128 * 128 * 2);
  _Float16* p_cw   = (_Float16*)alloc(128 * 128 * 2);

  auto pack = [&](const float* B, int K, int Nw, int ld, int tr, _Float16* o) {
    int total = (K / 32) * (Nw / 16) * 32;
    k_pack_b<<<(total + 255) / 256, 256, 0, stream>>>(B, K, Nw, ld, tr, o);
  };
  pack(m1w1, 768, 128, 128, 0, p_m1w1);
  pack(m1w2, 128, 128, 128, 0, p_m1w2);
  pack(g1w,  128, 256, 256, 0, p_g1w);
  pack(g2w,  256, 256, 256, 0, p_g2w);
  pack(m2w1, 128, 128, 128, 0, p_m2w1);
  pack(m2w2, 128, 128, 128, 0, p_m2w2);
  pack(cw,   128, 128, 128, 1, p_cw);   // einsum 'ntd,od': B[k=d][n=o] = cw[o][d]

  auto gemm = [&](const float* A, int lda, int M, int K, const _Float16* Bp, int Nw,
                  const float* bias, float* C, int ldc, int coff, int act) {
    dim3 grid((unsigned)((M + 127) / 128), (unsigned)(Nw / 128));
    if (act)
      k_gemm<1, 8><<<grid, 256, 0, stream>>>(A, lda, M, K, (const v16h*)Bp, Nw / 16, bias, C, ldc, coff);
    else
      k_gemm<0, 8><<<grid, 256, 0, stream>>>(A, lda, M, K, (const v16h*)Bp, Nw / 16, bias, C, ldc, coff);
  };

  // 1) embedding gather
  k_gather_emb<<<(NN * FF * EMBD + 255) / 256, 256, 0, stream>>>(sf, tables, B0);

  // 2) MLP1
  gemm(B0, 768, NN, 768, p_m1w1, 128, m1b1, B1, 128, 0, 1);
  gemm(B1, 128, NN, 128, p_m1w2, 128, m1b2, B2, 128, 0, 0);  // h1

  // 3) GAT layer 1
  gemm(B2, 128, NN, 128, p_g1w, 256, nullptr, B3, 256, 0, 0);  // xh1
  k_attn<<<(NN * 2 + 255) / 256, 256, 0, stream>>>(B3, g1as, g1ad, aS, aD, menc);
  k_edge_max<<<(EE * 2 + 255) / 256, 256, 0, stream>>>(src, dst, aS, aD, menc);
  k_node_z<<<(NN * 2 + 255) / 256, 256, 0, stream>>>(aS, aD, menc, zz);
  k_edge_z<<<(EE * 2 + 255) / 256, 256, 0, stream>>>(src, dst, aS, aD, menc, zz);
  k_agg_init<<<(NN * 256 + 255) / 256, 256, 0, stream>>>(B3, aS, aD, menc, zz, B1);
  k_edge_agg<<<(EE * 8 + 255) / 256, 256, 0, stream>>>(src, dst, B3, aS, aD, menc, zz, B1);
  k_elu_bias<<<(NN * 256 + 255) / 256, 256, 0, stream>>>(B1, g1b, B0);  // h2

  // 4) GAT layer 2
  gemm(B0, 256, NN, 256, p_g2w, 256, nullptr, B3, 256, 0, 0);  // xh2
  k_attn<<<(NN * 2 + 255) / 256, 256, 0, stream>>>(B3, g2as, g2ad, aS, aD, menc);
  k_edge_max<<<(EE * 2 + 255) / 256, 256, 0, stream>>>(src, dst, aS, aD, menc);
  k_node_z<<<(NN * 2 + 255) / 256, 256, 0, stream>>>(aS, aD, menc, zz);
  k_edge_z<<<(EE * 2 + 255) / 256, 256, 0, stream>>>(src, dst, aS, aD, menc, zz);
  k_agg_init<<<(NN * 256 + 255) / 256, 256, 0, stream>>>(B3, aS, aD, menc, zz, B1);
  k_edge_agg<<<(EE * 8 + 255) / 256, 256, 0, stream>>>(src, dst, B3, aS, aD, menc, zz, B1);
  k_mean_bias<<<(NN * 128 + 255) / 256, 256, 0, stream>>>(B1, g2b, B2);  // h3

  // 5) MLP2
  gemm(B2, 128, NN, 128, p_m2w1, 128, m2b1, B0, 128, 0, 1);
  gemm(B0, 128, NN, 128, p_m2w2, 128, m2b2, B1, 128, 0, 0);  // static

  // 6) output: static half, conv half fused into d_out, special rows
  k_broadcast_static<<<(NN * TT * 32 + 255) / 256, 256, 0, stream>>>(B1, out);
  gemm(dynamic, 128, NN * TT, 128, p_cw, 128, cb, out, 256, 128, 0);
  k_special<<<(4 * TT * 256 + 255) / 256, 256, 0, stream>>>(out);
}
